// MultiHeadAttention_31009663877371
// MI455X (gfx1250) — matrix operand order repaired and checked by reading, not compiled, not run on hardware
//
#include <hip/hip_runtime.h>
#include <hip/hip_bf16.h>

typedef unsigned short ushort_t;
typedef __attribute__((ext_vector_type(16))) __bf16        v16bf;
typedef __attribute__((ext_vector_type(8)))  float         v8f;
typedef __attribute__((ext_vector_type(4)))  unsigned int  v4u;
typedef __attribute__((ext_vector_type(4)))  float         v4f;

#define S_SEQ 2048
#define EMBED 1024
#define HEADS 16
#define HD    64
#define BATCH 4
#define MROWS (BATCH * S_SEQ) /* 8192 */

// scores[16][2048] f32 + red[16][16] + rstat[16] + part[4][16][16]
#define ATTN_SMEM (16 * 2048 * 4 + 16 * 16 * 4 + 16 * 4 + 4 * 16 * 16 * 4)

union FragB16 {
    v16bf    v;
    v4u      q[2];
    ushort_t u[16];
};

__device__ __forceinline__ ushort_t f32_to_bf16(float x) {
    unsigned u = __float_as_uint(x);
    if ((u & 0x7F800000u) == 0x7F800000u) return (ushort_t)(u >> 16); // inf/nan: truncate
    return (ushort_t)((u + 0x7FFFu + ((u >> 16) & 1u)) >> 16);       // round-nearest-even
}

__device__ __forceinline__ v8f wmma_bf16(const FragB16& a, const FragB16& b, v8f c) {
    return __builtin_amdgcn_wmma_f32_16x16x32_bf16(
        /*neg_a=*/false, a.v, /*neg_b=*/false, b.v,
        /*c_mod=*/(short)0, c, /*reuse_a=*/false, /*reuse_b=*/false);
}

// A-matrix 16x32 bf16 tile at (m0,k0) of row-major [*,ldk]:
// lane L: M = L&15 ; elements 0..7 -> K = base..base+7, 8..15 -> K = base+16..base+23,
// base = (L>=16)?8:0   (per CDNA5 ISA 16-bit A layout)
__device__ __forceinline__ FragB16 load_a_frag(const ushort_t* base, int ldk,
                                               int m0, int k0, int lane) {
    int m  = m0 + (lane & 15);
    int kb = k0 + ((lane >> 4) << 3);
    const ushort_t* p = base + (size_t)m * ldk + kb;
    FragB16 f;
    f.q[0] = *(const v4u*)(p);
    f.q[1] = *(const v4u*)(p + 16);
    return f;
}

// B-matrix 32x16 tile where B[k][n] = Wrow[n][k] (W row-major [*,ldk]):
// lane L: N = L&15 ; elements 0..7 -> K = base..base+7, 8..15 -> K = base+16..base+23, base = k0 + ((L>=16)?8:0)
__device__ __forceinline__ FragB16 load_b_frag(const ushort_t* base, int ldk,
                                               int n0, int k0, int lane) {
    int n  = n0 + (lane & 15);
    int kb = k0 + ((lane >> 4) << 3);
    const ushort_t* p = base + (size_t)n * ldk + kb;
    FragB16 f;
    f.q[0] = *(const v4u*)(p);
    f.q[1] = *(const v4u*)(p + 16);
    return f;
}

__global__ __launch_bounds__(256) void cvt_bf16_kernel(const float* __restrict__ src,
                                                       ushort_t* __restrict__ dst, int n) {
    int i = blockIdx.x * 256 + threadIdx.x;
    if (i < n) dst[i] = f32_to_bf16(src[i]);
}

// C[m,n] = sum_k A[m,k] * W[n,k] + bias[n].  M=8192, N=1024, K=1024.
// Wave tile 64x32 (4 M-tiles x 2 N-tiles): 12 b128 loads per 8 WMMAs.
// Block = 8 waves (2x4) covering 128x128.
// mode 0: bf16 out at [B,H,S,hd] ; mode 1: bf16 out at [B,H,hd,S] ; mode 2: f32 out [M,N]
__global__ __launch_bounds__(256) void gemm_bf16_kernel(
    const ushort_t* __restrict__ A, const ushort_t* __restrict__ W,
    const float* __restrict__ bias, ushort_t* __restrict__ outb,
    float* __restrict__ outf, int mode)
{
    const int lane = threadIdx.x & 31;
    const int w    = threadIdx.x >> 5;
    const int m0   = blockIdx.y * 128 + (w >> 2) * 64;
    const int n0   = blockIdx.x * 128 + (w & 3) * 32;

    v8f acc[4][2];
#pragma unroll
    for (int i = 0; i < 4; ++i)
#pragma unroll
        for (int j = 0; j < 2; ++j) acc[i][j] = (v8f){};

    for (int kk = 0; kk < EMBED; kk += 32) {
        FragB16 a[4], b[2];
#pragma unroll
        for (int i = 0; i < 4; ++i)
            a[i] = load_a_frag(A, EMBED, m0 + i * 16, kk, lane);
        b[0] = load_b_frag(W, EMBED, n0,      kk, lane);
        b[1] = load_b_frag(W, EMBED, n0 + 16, kk, lane);
        if (kk + 32 < EMBED) {
            __builtin_prefetch(A + (size_t)(m0 + (lane & 15)) * EMBED + kk + 32, 0, 1);
            __builtin_prefetch(W + (size_t)(n0 + (lane & 15)) * EMBED + kk + 32, 0, 1);
        }
#pragma unroll
        for (int i = 0; i < 4; ++i)
#pragma unroll
            for (int j = 0; j < 2; ++j)
                acc[i][j] = wmma_bf16(a[i], b[j], acc[i][j]);
    }

    const int mb = (lane >> 4) << 3; // C/D layout: lanes>=16 hold M = r + 8
    const int nl = lane & 15;
#pragma unroll
    for (int mi = 0; mi < 4; ++mi) {
#pragma unroll
        for (int ni = 0; ni < 2; ++ni) {
#pragma unroll
            for (int r = 0; r < 8; ++r) {
                int m = m0 + mi * 16 + mb + r;
                int n = n0 + ni * 16 + nl;
                float val = acc[mi][ni][r] + bias[n];
                if (mode == 2) {
                    outf[(size_t)m * EMBED + n] = val;
                } else {
                    int b = m >> 11, s = m & (S_SEQ - 1);
                    int h = n >> 6,  d = n & (HD - 1);
                    size_t addr = (mode == 0)
                        ? ((size_t)(b * HEADS + h) * S_SEQ + s) * HD + d
                        : ((size_t)(b * HEADS + h) * HD + d) * S_SEQ + s;
                    outb[addr] = f32_to_bf16(val);
                }
            }
        }
    }
}

// One block = (b,h, 16 query rows). Scores [16][2048] f32 staged in LDS (128 KB of
// the 320 KB WGP LDS), full softmax, then P@V with split-K across wave pairs.
__global__ __launch_bounds__(256) void attn_kernel(
    const ushort_t* __restrict__ qg, const ushort_t* __restrict__ kg,
    const ushort_t* __restrict__ vtg, ushort_t* __restrict__ ao)
{
    extern __shared__ char smem[];
    float* sc    = (float*)smem;                    // [16][2048]
    float* red   = (float*)(smem + 16 * 2048 * 4);  // [16][16]
    float* rstat = red + 16 * 16;                   // [16] = 1/rowsum
    float* part  = rstat + 16;                      // [4][16][16]

    const int t = threadIdx.x, lane = t & 31, w = t >> 5;
    const int bh = blockIdx.y;
    const int q0 = blockIdx.x * 16;
    const ushort_t* qb  = qg  + ((size_t)bh * S_SEQ + q0) * HD;
    const ushort_t* kb  = kg  + (size_t)bh * S_SEQ * HD;
    const ushort_t* vtb = vtg + (size_t)bh * HD * S_SEQ;

    const int mb = (lane >> 4) << 3;
    const int nl = lane & 15;

    // Q fragments for the two hd k-steps (resident for whole phase 1)
    FragB16 aq0 = load_a_frag(qb, HD, 0, 0,  lane);
    FragB16 aq1 = load_a_frag(qb, HD, 0, 32, lane);

    // ---- phase 1: S = (Q @ K^T) / sqrt(hd), 8 waves x 16 n-tiles
    for (int nt = 0; nt < 16; ++nt) {
        int n0 = (w * 16 + nt) * 16;
        FragB16 b0 = load_b_frag(kb, HD, n0, 0,  lane);
        FragB16 b1 = load_b_frag(kb, HD, n0, 32, lane);
        v8f acc = {};
        acc = wmma_bf16(aq0, b0, acc);
        acc = wmma_bf16(aq1, b1, acc);
#pragma unroll
        for (int r = 0; r < 8; ++r)
            sc[(mb + r) * S_SEQ + n0 + nl] = acc[r] * 0.125f; // 1/sqrt(64)
    }
    __syncthreads();

    // ---- phase 2: softmax, 16 threads per row, 128 cols each
    {
        int row = t >> 4, seg = t & 15;
        float* rp = sc + row * S_SEQ + seg * 128;
        float mx = -3.4e38f;
        for (int i = 0; i < 128; ++i) mx = fmaxf(mx, rp[i]);
        red[row * 16 + seg] = mx;
        __syncthreads();
        float rmx = red[row * 16];
        for (int i = 1; i < 16; ++i) rmx = fmaxf(rmx, red[row * 16 + i]);
        __syncthreads();
        float sum = 0.f;
        for (int i = 0; i < 128; ++i) { float e = __expf(rp[i] - rmx); rp[i] = e; sum += e; }
        red[row * 16 + seg] = sum;
        __syncthreads();
        if (seg == 0) {
            float s = 0.f;
            for (int i = 0; i < 16; ++i) s += red[row * 16 + i];
            rstat[row] = 1.0f / s;
        }
    }
    __syncthreads();

    // ---- phase 3: O = P @ V ; 8 waves = 4 hd n-tiles x 2 K-halves of 1024
    const int nt = w & 3, khalf = w >> 2;
    v8f acc = {};
    const float rs = rstat[nl]; // lane's A-fragment row is nl
    for (int step = 0; step < 32; ++step) {
        int k0 = khalf * 1024 + step * 32;
        const float* prow = sc + (size_t)nl * S_SEQ + k0 + mb;
        v4f f0 = *(const v4f*)(prow);
        v4f f1 = *(const v4f*)(prow + 4);
        v4f f2 = *(const v4f*)(prow + 16);
        v4f f3 = *(const v4f*)(prow + 20);
        FragB16 a;
#pragma unroll
        for (int i = 0; i < 4; ++i) {
            a.u[i]      = f32_to_bf16(f0[i] * rs);
            a.u[4 + i]  = f32_to_bf16(f1[i] * rs);
            a.u[8 + i]  = f32_to_bf16(f2[i] * rs);
            a.u[12 + i] = f32_to_bf16(f3[i] * rs);
        }
        FragB16 bv = load_b_frag(vtb, S_SEQ, nt * 16, k0, lane); // V^T rows contiguous
        acc = wmma_bf16(a, bv, acc);
    }
    if (khalf == 1) {
#pragma unroll
        for (int r = 0; r < 8; ++r)
            part[(nt * 16 + mb + r) * 16 + nl] = acc[r];
    }
    __syncthreads();
    if (khalf == 0) {
        const int b = bh >> 4, h = bh & 15;
#pragma unroll
        for (int r = 0; r < 8; ++r) {
            int M = mb + r;
            float val = acc[r] + part[(nt * 16 + M) * 16 + nl];
            size_t row = (size_t)b * S_SEQ + q0 + M;
            int col = h * HD + nt * 16 + nl;
            ao[row * EMBED + col] = f32_to_bf16(val); // [B,S,H,hd] == [M, D]
        }
    }
}

__global__ __launch_bounds__(256) void layernorm_kernel(
    float* __restrict__ out, const float* __restrict__ lw, const float* __restrict__ lb)
{
    __shared__ float s1[256], s2[256];
    const int t = threadIdx.x;
    float* p = out + (size_t)blockIdx.x * EMBED;
    float v[4]; float sum = 0.f, sq = 0.f;
#pragma unroll
    for (int i = 0; i < 4; ++i) {
        v[i] = p[t + i * 256];
        sum += v[i]; sq += v[i] * v[i];
    }
    s1[t] = sum; s2[t] = sq;
    __syncthreads();
    for (int off = 128; off > 0; off >>= 1) {
        if (t < off) { s1[t] += s1[t + off]; s2[t] += s2[t + off]; }
        __syncthreads();
    }
    float mu  = s1[0] * (1.0f / EMBED);
    float var = s2[0] * (1.0f / EMBED) - mu * mu;
    float inv = rsqrtf(var + 1e-5f);
#pragma unroll
    for (int i = 0; i < 4; ++i) {
        int c = t + i * 256;
        p[c] = (v[i] - mu) * inv * lw[c] + lb[c];
    }
}

extern "C" void kernel_launch(void* const* d_in, const int* in_sizes, int n_in,
                              void* d_out, int out_size, void* d_ws, size_t ws_size,
                              hipStream_t stream)
{
    (void)in_sizes; (void)n_in; (void)out_size; (void)ws_size;
    const float* x   = (const float*)d_in[0];
    const float* Wq  = (const float*)d_in[1];
    const float* bq  = (const float*)d_in[2];
    const float* Wk  = (const float*)d_in[3];
    const float* bk  = (const float*)d_in[4];
    const float* Wv  = (const float*)d_in[5];
    const float* bv  = (const float*)d_in[6];
    const float* Wo  = (const float*)d_in[7];
    const float* bo  = (const float*)d_in[8];
    const float* lnw = (const float*)d_in[9];
    const float* lnb = (const float*)d_in[10];

    char* ws = (char*)d_ws;
    const size_t MB = (size_t)1 << 20;
    ushort_t* xb  = (ushort_t*)(ws);            // 16 MB, recycled as `ao` after QKV
    ushort_t* wqb = (ushort_t*)(ws + 16 * MB);  // 2 MB each
    ushort_t* wkb = (ushort_t*)(ws + 18 * MB);
    ushort_t* wvb = (ushort_t*)(ws + 20 * MB);
    ushort_t* wob = (ushort_t*)(ws + 22 * MB);
    ushort_t* qb  = (ushort_t*)(ws + 24 * MB);  // [B,H,S,hd] bf16, 16 MB
    ushort_t* kb  = (ushort_t*)(ws + 40 * MB);  // [B,H,S,hd]
    ushort_t* vtb = (ushort_t*)(ws + 56 * MB);  // [B,H,hd,S]  -> 72 MB total
    ushort_t* ao  = xb;

    (void)hipFuncSetAttribute(reinterpret_cast<const void*>(attn_kernel),
                              hipFuncAttributeMaxDynamicSharedMemorySize, ATTN_SMEM);

    // fp32 -> bf16 staging
    cvt_bf16_kernel<<<(MROWS * EMBED) / 256, 256, 0, stream>>>(x, xb, MROWS * EMBED);
    cvt_bf16_kernel<<<(EMBED * EMBED) / 256, 256, 0, stream>>>(Wq, wqb, EMBED * EMBED);
    cvt_bf16_kernel<<<(EMBED * EMBED) / 256, 256, 0, stream>>>(Wk, wkb, EMBED * EMBED);
    cvt_bf16_kernel<<<(EMBED * EMBED) / 256, 256, 0, stream>>>(Wv, wvb, EMBED * EMBED);
    cvt_bf16_kernel<<<(EMBED * EMBED) / 256, 256, 0, stream>>>(Wo, wob, EMBED * EMBED);

    dim3 gg(EMBED / 128, MROWS / 128);
    gemm_bf16_kernel<<<gg, 256, 0, stream>>>(xb, wqb, bq, qb,  nullptr, 0);
    gemm_bf16_kernel<<<gg, 256, 0, stream>>>(xb, wkb, bk, kb,  nullptr, 0);
    gemm_bf16_kernel<<<gg, 256, 0, stream>>>(xb, wvb, bv, vtb, nullptr, 1);

    attn_kernel<<<dim3(S_SEQ / 16, BATCH * HEADS), 256, ATTN_SMEM, stream>>>(qb, kb, vtb, ao);

    gemm_bf16_kernel<<<gg, 256, 0, stream>>>(ao, wob, bo, nullptr, (float*)d_out, 2);
    layernorm_kernel<<<MROWS, 256, 0, stream>>>((float*)d_out, lnw, lnb);
}
